// MultiHeadAttention_6622839570945
// MI455X (gfx1250) — matrix operand order repaired and checked by reading, not compiled, not run on hardware
//
#include <hip/hip_runtime.h>
#include <hip/hip_bf16.h>

// ---------------------------------------------------------------------------
// MI455X (gfx1250) multi-head attention, f16 WMMA with f32 accumulation.
// B=2, N=2048, C=1024, H=16, D=64.
// Entire working set (~50MB) is L2-resident (192MB), so the kernel is
// matrix-core bound: every GEMM runs on v_wmma_f32_16x16x32_f16.
// ---------------------------------------------------------------------------

typedef __attribute__((ext_vector_type(16))) _Float16 v16h;
typedef __attribute__((ext_vector_type(8)))  _Float16 v8h;
typedef __attribute__((ext_vector_type(8)))  float    v8f;

#define WMMA16(a, b, c) \
  __builtin_amdgcn_wmma_f32_16x16x32_f16(false, (a), false, (b), (short)0, (c), false, false)

__device__ __forceinline__ v8f zero_v8f() {
  v8f z;
#pragma unroll
  for (int i = 0; i < 8; ++i) z[i] = 0.0f;
  return z;
}

// A fragment (16x32, f16), A row-major [.. x lda], tile at (row0, k0).
// ISA layout: lane<16 -> row=row0+lane, elems 0..7 = K k0..k0+7, elems 8..15 = K k0+16..23
//             lane>=16 -> same row, K +8..15 and +24..31.
__device__ __forceinline__ v16h load_frag_a(const _Float16* __restrict__ A, int lda,
                                            int row0, int k0, int lane) {
  const int half = lane >> 4;
  const int r    = lane & 15;
  const _Float16* p = A + (size_t)(row0 + r) * lda + k0 + half * 8;
  union { v16h v; v8h h[2]; } u;
  u.h[0] = *(const v8h*)(p);        // 16B aligned
  u.h[1] = *(const v8h*)(p + 16);   // 16B aligned
  return u.v;
}

// B fragment (32x16, f16) for C[m,n] = sum_k A[m,k]*Bm[n,k]  (Bm row-major [N x ldb]).
// ISA layout: lane<16 -> col=col0+lane, elems 0..7 = K k0..k0+7, elems 8..15 = K k0+16..23
//             lane>=16 -> col=col0+lane-16, K +8..15 and +24..31.
__device__ __forceinline__ v16h load_frag_b(const _Float16* __restrict__ Bm, int ldb,
                                            int col0, int k0, int lane) {
  const int half = lane >> 4;
  const int c    = lane & 15;
  const _Float16* p = Bm + (size_t)(col0 + c) * ldb + k0 + half * 8;
  union { v16h v; v8h h[2]; } u;
  u.h[0] = *(const v8h*)(p);        // 16B aligned
  u.h[1] = *(const v8h*)(p + 16);   // 16B aligned
  return u.v;
}

// ---------------------------------------------------------------------------
// f32 -> f16 conversion
// ---------------------------------------------------------------------------
__global__ void cvt_f32_to_f16(const float* __restrict__ src, _Float16* __restrict__ dst,
                               size_t n) {
  size_t i = (size_t)blockIdx.x * blockDim.x + threadIdx.x;
  size_t stride = (size_t)gridDim.x * blockDim.x;
  for (; i < n; i += stride) dst[i] = (_Float16)src[i];
}

// ---------------------------------------------------------------------------
// Tiled GEMM: C[M,N] = A[M,K] @ Bm[N,K]^T + bias[N]
// mode 0: write f32 C (ldc) ;  mode 1: scatter f16 into per-head Q, K, V^T.
// Block = 256 threads = 8 waves (4 M x 2 N), wave tile 32x64, block tile 128x128.
// ---------------------------------------------------------------------------
__global__ __launch_bounds__(256)
void gemm_f16_wmma(const _Float16* __restrict__ A, const _Float16* __restrict__ Bm,
                   const float* __restrict__ bias,
                   int M, int N, int K, int mode,
                   float* __restrict__ Cout, int ldc,
                   _Float16* __restrict__ qh, _Float16* __restrict__ kh,
                   _Float16* __restrict__ vth) {
  const int lane = threadIdx.x & 31;
  const int wave = threadIdx.x >> 5;
  const int wy = wave >> 1;                       // 0..3
  const int wx = wave & 1;                        // 0..1
  const int bm0 = blockIdx.y * 128 + wy * 32;     // wave M origin
  const int bn0 = blockIdx.x * 128 + wx * 64;     // wave N origin

  v8f acc[2][4];
#pragma unroll
  for (int i = 0; i < 2; ++i)
#pragma unroll
    for (int j = 0; j < 4; ++j) acc[i][j] = zero_v8f();

  for (int k0 = 0; k0 < K; k0 += 32) {
    if (k0 + 32 < K) {  // global_prefetch_b8 of next K slab
      __builtin_prefetch((const void*)(A + (size_t)bm0 * K + k0 + 32), 0, 1);
      __builtin_prefetch((const void*)(Bm + (size_t)bn0 * K + k0 + 32), 0, 1);
    }
    v16h afr[2];
#pragma unroll
    for (int i = 0; i < 2; ++i) afr[i] = load_frag_a(A, K, bm0 + 16 * i, k0, lane);
    v16h bfr[4];
#pragma unroll
    for (int j = 0; j < 4; ++j) bfr[j] = load_frag_b(Bm, K, bn0 + 16 * j, k0, lane);
#pragma unroll
    for (int i = 0; i < 2; ++i)
#pragma unroll
      for (int j = 0; j < 4; ++j) acc[i][j] = WMMA16(afr[i], bfr[j], acc[i][j]);
  }

  // Epilogue. C-tile layout: VGPR r, lane<16 -> (row r,  col lane)
  //                                   lane>=16 -> (row r+8, col lane-16)
  const int half = lane >> 4;
  const int c    = lane & 15;
  const int Nseq = 2048, Dh = 64, Hh = 16;
#pragma unroll
  for (int i = 0; i < 2; ++i) {
#pragma unroll
    for (int j = 0; j < 4; ++j) {
#pragma unroll
      for (int r = 0; r < 8; ++r) {
        const int gr = bm0 + 16 * i + r + half * 8;   // global row (b*N + n)
        const int gc = bn0 + 16 * j + c;              // global col
        float v = acc[i][j][r] + bias[gc];
        if (mode == 0) {
          Cout[(size_t)gr * ldc + gc] = v;
        } else {
          const int b  = gr >> 11;                    // / 2048
          const int n  = gr & 2047;
          const int wh = gc >> 10;                    // 0:Q 1:K 2:V
          const int cc = gc & 1023;
          const int h  = cc >> 6;
          const int d  = cc & 63;
          const _Float16 hv = (_Float16)v;
          const int bh = b * Hh + h;
          if (wh == 0)       qh[((size_t)bh * Nseq + n) * Dh + d] = hv;
          else if (wh == 1)  kh[((size_t)bh * Nseq + n) * Dh + d] = hv;
          else               vth[((size_t)bh * Dh + d) * Nseq + n] = hv;  // V^T
        }
      }
    }
  }
}

// ---------------------------------------------------------------------------
// Flash attention. One block = 128 threads = 4 waves; each wave owns 16 query
// rows. Scores are computed TRANSPOSED (S^T = K @ Q^T) so the f32 C-tile of
// exp(S) maps 1:1 onto the f16 A-fragment layout of the P @ V WMMA — the P
// fragment is rebuilt purely in registers (no LDS transpose needed).
// ---------------------------------------------------------------------------
__global__ __launch_bounds__(128)
void attn_flash_wmma(const _Float16* __restrict__ qg, const _Float16* __restrict__ kg,
                     const _Float16* __restrict__ vtg, _Float16* __restrict__ oh) {
  const int Nseq = 2048, Dh = 64, Cc = 1024;
  const int lane = threadIdx.x & 31;
  const int wave = threadIdx.x >> 5;
  const int bh = blockIdx.x >> 5;              // 32 row-blocks per (b,h)
  const int rb = blockIdx.x & 31;
  const int b  = bh >> 4;
  const int h  = bh & 15;

  const _Float16* Q  = qg  + (size_t)bh * Nseq * Dh;
  const _Float16* Kp = kg  + (size_t)bh * Nseq * Dh;
  const _Float16* Vt = vtg + (size_t)bh * Dh * Nseq;   // [D x N]

  const int m0 = rb * 64 + wave * 16;          // this wave's query rows

  // Q^T is the B operand of the score WMMA (columns = query rows). Load once.
  const v16h bq0 = load_frag_b(Q, Dh, m0, 0, lane);
  const v16h bq1 = load_frag_b(Q, Dh, m0, 32, lane);

  // Per-lane softmax state for query row (lane & 15); lanes n and n+16 share a row.
  float mrow = -__builtin_inff();
  float lrow = 0.0f;
  v8f oacc[4];
#pragma unroll
  for (int j = 0; j < 4; ++j) oacc[j] = zero_v8f();

  const float sc = 0.125f * 1.44269504088896340736f;   // 1/sqrt(64) * log2(e)
  const int half = lane >> 4;

  for (int kb = 0; kb < Nseq; kb += 32) {
    // --- S^T tiles: t0 = keys kb..kb+15, t1 = keys kb+16..kb+31 ----------
    v8f s0 = zero_v8f(), s1 = zero_v8f();
    {
      v16h ka = load_frag_a(Kp, Dh, kb, 0, lane);
      v16h kc = load_frag_a(Kp, Dh, kb, 32, lane);
      s0 = WMMA16(ka, bq0, s0);
      s0 = WMMA16(kc, bq1, s0);
      ka = load_frag_a(Kp, Dh, kb + 16, 0, lane);
      kc = load_frag_a(Kp, Dh, kb + 16, 32, lane);
      s1 = WMMA16(ka, bq0, s1);
      s1 = WMMA16(kc, bq1, s1);
    }

    // scale into log2 domain; per-lane row max over this lane's 16 keys
    float mx = -__builtin_inff();
#pragma unroll
    for (int r = 0; r < 8; ++r) {
      s0[r] *= sc;  s1[r] *= sc;
      mx = fmaxf(mx, fmaxf(s0[r], s1[r]));
    }
    mx = fmaxf(mx, __shfl_xor(mx, 16, 32));     // merge the two lane-halves of the row
    const float mnew = fmaxf(mrow, mx);
    const float corr = exp2f(mrow - mnew);      // 0 on first block
    mrow = mnew;

    // exp(P) and in-register A-fragment: elems 0..7 <- t0, elems 8..15 <- t1
    float ls = 0.0f;
    v16h pa;
#pragma unroll
    for (int r = 0; r < 8; ++r) {
      const float p0 = exp2f(s0[r] - mnew);
      const float p1 = exp2f(s1[r] - mnew);
      ls += p0 + p1;
      pa[r]     = (_Float16)p0;
      pa[8 + r] = (_Float16)p1;
    }
    ls += __shfl_xor(ls, 16, 32);
    lrow = lrow * corr + ls;

    // rescale O accumulators: broadcast corr from row-owner lane (half*8 + r)
#pragma unroll
    for (int r = 0; r < 8; ++r) {
      const float cr = __shfl(corr, half * 8 + r, 32);
#pragma unroll
      for (int j = 0; j < 4; ++j) oacc[j][r] *= cr;
    }

    // O += P @ V ; B operand from V^T (contiguous in key index)
#pragma unroll
    for (int j = 0; j < 4; ++j) {
      const v16h vb = load_frag_b(Vt, Nseq, j * 16, kb, lane);
      oacc[j] = WMMA16(pa, vb, oacc[j]);
    }
  }

  // Finalize: divide by row sums, store f16 to oh[b*N + q, h*64 + d]
  const float inv = 1.0f / lrow;
  const int c = lane & 15;
#pragma unroll
  for (int r = 0; r < 8; ++r) {
    const float ir = __shfl(inv, half * 8 + r, 32);
    const int gr = m0 + r + half * 8;
#pragma unroll
    for (int j = 0; j < 4; ++j) {
      const int gc = h * 64 + j * 16 + c;
      oh[((size_t)b * Nseq + gr) * Cc + gc] = (_Float16)(oacc[j][r] * ir);
    }
  }
}

// ---------------------------------------------------------------------------
// Host-side launcher
// ---------------------------------------------------------------------------
extern "C" void kernel_launch(void* const* d_in, const int* in_sizes, int n_in,
                              void* d_out, int out_size, void* d_ws, size_t ws_size,
                              hipStream_t stream) {
  (void)in_sizes; (void)n_in; (void)out_size; (void)ws_size;
  const int B = 2, Nseq = 2048, C = 1024, H = 16, Dh = 64;
  const int M = B * Nseq;  // 4096

  const float* x      = (const float*)d_in[0];
  const float* qkv_w  = (const float*)d_in[1];
  const float* qkv_b  = (const float*)d_in[2];
  const float* out_w  = (const float*)d_in[3];
  const float* out_b  = (const float*)d_in[4];
  float*       out    = (float*)d_out;

  // Workspace carve-up (all f16). Total 40 MB (oh reuses xh region).
  char* ws = (char*)d_ws;
  _Float16* xh   = (_Float16*)ws;  ws += (size_t)M * C * 2;          //  8 MB
  _Float16* wqkv = (_Float16*)ws;  ws += (size_t)3 * C * C * 2;      //  6 MB
  _Float16* wout = (_Float16*)ws;  ws += (size_t)C * C * 2;          //  2 MB
  _Float16* qh   = (_Float16*)ws;  ws += (size_t)B * H * Nseq * Dh * 2;  // 8 MB
  _Float16* kh   = (_Float16*)ws;  ws += (size_t)B * H * Nseq * Dh * 2;  // 8 MB
  _Float16* vth  = (_Float16*)ws;  ws += (size_t)B * H * Dh * Nseq * 2;  // 8 MB
  _Float16* oh   = xh;  // x no longer needed once QKV projection is done

  // 1) f32 -> f16 conversions
  cvt_f32_to_f16<<<1024, 256, 0, stream>>>(x, xh, (size_t)M * C);
  cvt_f32_to_f16<<<1024, 256, 0, stream>>>(qkv_w, wqkv, (size_t)3 * C * C);
  cvt_f32_to_f16<<<512, 256, 0, stream>>>(out_w, wout, (size_t)C * C);

  // 2) QKV projection, scatter into Q / K / V^T head layouts
  {
    dim3 grid(3 * C / 128, M / 128);  // 24 x 32
    gemm_f16_wmma<<<grid, 256, 0, stream>>>(xh, wqkv, qkv_b, M, 3 * C, C,
                                            /*mode=*/1, nullptr, 0, qh, kh, vth);
  }

  // 3) Flash attention per (b, h, 64-row block)
  attn_flash_wmma<<<B * H * (Nseq / 64), 128, 0, stream>>>(qh, kh, vth, oh);

  // 4) Output projection -> f32 out
  {
    dim3 grid(C / 128, M / 128);  // 8 x 32
    gemm_f16_wmma<<<grid, 256, 0, stream>>>(oh, wout, out_b, M, C, C,
                                            /*mode=*/0, out, C, nullptr, nullptr, nullptr);
  }
}
